// PI_GCN_69973607186732
// MI455X (gfx1250) — matrix operand order repaired and checked by reading, not compiled, not run on hardware
//
#include <hip/hip_runtime.h>

#define N_NODES 30000
#define N_EDGES 80000
#define EH      (N_EDGES/2)
#define I_LAYERS 3
#define N_ITER   3
#define R_ITER   2               // setup_inputs fixes r_iter = 2
#define K_OUTER  (N_ITER + R_ITER)
#define J_HEADS  (I_LAYERS * N_ITER)
#define HW_EXP   1.852f
#define INV_HW   (1.0f/1.852f)
#define ZETA_F   1e-32f
#define NEG_F    (-1e30f)
#define IMIN     ((int)0x80000000)

typedef __attribute__((ext_vector_type(16))) _Float16 v16h;
typedef __attribute__((ext_vector_type(8)))  _Float16 v8h;
typedef __attribute__((ext_vector_type(8)))  float    v8f;

// ---------------- device helpers ----------------
// branchless SELU: exp always evaluated on min(x,0), select via cndmask
__device__ __forceinline__ float selu_f(float x) {
    const float l  = 1.0507009873554805f;
    const float la = 1.7580993408473766f;            // l * alpha
    float e = __expf(fminf(x, 0.f)) - 1.f;
    return x > 0.f ? l * x : la * e;
}
// monotone float<->int encoding so int atomicMax == float max
__device__ __forceinline__ int fenc(float f) {
    int i = __float_as_int(f);
    return i >= 0 ? i : (i ^ 0x7fffffff);
}
__device__ __forceinline__ float fdec(int k) {
    int i = k >= 0 ? k : (k ^ 0x7fffffff);
    return __int_as_float(i);
}

__device__ __forceinline__ v8f wmma_f16(v16h a, v16h b, v8f c) {
    return __builtin_amdgcn_wmma_f32_16x16x32_f16(
        false, a, false, b, (short)0, c, false, false);
}

// A fragment: 16x32 f16 tile, row m = lane&15.  Per the ISA layout each lane's
// 16 halves are two contiguous 8-half runs: [hi*8 .. hi*8+7] and [+16 .. +23].
__device__ __forceinline__ v16h load_a_frag(const _Float16* base, int lda, int lane) {
    const _Float16* p = base + (lane & 15) * lda + ((lane >> 4) << 3);
    v8h lo = *(const v8h*)(p);
    v8h hi = *(const v8h*)(p + 16);
    return __builtin_shufflevector(lo, hi, 0,1,2,3,4,5,6,7,8,9,10,11,12,13,14,15);
}
// B fragment: pre-swizzled fragment-major weights -> one v16h (32B) load/lane.
__device__ __forceinline__ v16h load_b_frag(const _Float16* Wf, int kt, int nt, int lane) {
    return ((const v16h*)Wf)[(size_t)(kt * 8 + nt) * 32 + lane];
}

// ---------------- WMMA GEMM: gathered/concatenated A ----------------
// A row e = [S0[I0[e]] | S1[I1[e]] (| S2[I2[e]])], each segment 128 f16.
// C = A @ W (fragment-major f16), out = f16(selu(C)). block: 64 rows x 128 cols.
template <int NSEG>
__global__ void __launch_bounds__(256)
wmma_gemm_concat(const _Float16* __restrict__ S0, const int* __restrict__ I0,
                 const _Float16* __restrict__ S1, const int* __restrict__ I1,
                 const _Float16* __restrict__ S2, const int* __restrict__ I2,
                 const _Float16* __restrict__ Wf,
                 _Float16* __restrict__ outSel,
                 int rows) {
    constexpr int LDA  = NSEG * 128;
    constexpr int SEGV = NSEG * 16;                  // uint4 per row
    __shared__ uint4 As4[64 * SEGV];
    _Float16* As = (_Float16*)As4;
    const int tid   = threadIdx.x;
    const int row0g = blockIdx.x * 64;

    for (int v = tid; v < 64 * SEGV; v += 256) {
        int row = v / SEGV;
        int c   = v - row * SEGV;
        int seg = c >> 4;
        int off = c & 15;
        int e = row0g + row; if (e >= rows) e = rows - 1;
        const _Float16* S; const int* I;
        if (seg == 0)      { S = S0; I = I0; }
        else if (seg == 1) { S = S1; I = I1; }
        else               { S = S2; I = I2; }
        int idx = I ? I[e] : e;
        As4[v] = ((const uint4*)S)[(size_t)idx * 16 + off];
    }
    __syncthreads();

    const int lane = tid & 31;
    const int wave = tid >> 5;
    const int rt = wave & 3;                         // row tile 0..3
    const int ch = wave >> 2;                        // col half 0..1

    v8f acc0 = {}, acc1 = {}, acc2 = {}, acc3 = {};
#pragma unroll
    for (int kt = 0; kt < NSEG * 4; ++kt) {
        v16h a  = load_a_frag(As + rt * 16 * LDA + kt * 32, LDA, lane);
        acc0 = wmma_f16(a, load_b_frag(Wf, kt, ch * 4 + 0, lane), acc0);
        acc1 = wmma_f16(a, load_b_frag(Wf, kt, ch * 4 + 1, lane), acc1);
        acc2 = wmma_f16(a, load_b_frag(Wf, kt, ch * 4 + 2, lane), acc2);
        acc3 = wmma_f16(a, load_b_frag(Wf, kt, ch * 4 + 3, lane), acc3);
    }

    v8f accs[4] = {acc0, acc1, acc2, acc3};
    int mBase = row0g + rt * 16 + ((lane >> 4) << 3);    // VGPR v -> M = v + 8*hi
    int nb = lane & 15;
    if (row0g + 64 <= rows) {                            // full-tile fast path
#pragma unroll
        for (int j = 0; j < 4; ++j) {
            int col = (ch * 4 + j) * 16 + nb;
#pragma unroll
            for (int v = 0; v < 8; ++v)
                outSel[(size_t)(mBase + v) * 128 + col] = (_Float16)selu_f(accs[j][v]);
        }
    } else {
#pragma unroll
        for (int j = 0; j < 4; ++j) {
            int col = (ch * 4 + j) * 16 + nb;
#pragma unroll
            for (int v = 0; v < 8; ++v) {
                int row = mBase + v;
                if (row < rows)
                    outSel[(size_t)row * 128 + col] = (_Float16)selu_f(accs[j][v]);
            }
        }
    }
}

// ---------------- WMMA GEMM: plain A (rows x 128) -------------------
// out = f16(selu(A @ W)); if AGG also atomicMax segment-max scatter via rcvr.
template <bool AGG>
__global__ void __launch_bounds__(256)
wmma_gemm_plain(const _Float16* __restrict__ A,
                const _Float16* __restrict__ Wf,
                _Float16* __restrict__ outSel,
                int* __restrict__ aggEnc, const int* __restrict__ rcvr,
                int rows) {
    const int tid  = threadIdx.x;
    const int lane = tid & 31;
    const int wave = tid >> 5;
    const int rt = wave & 3, ch = wave >> 2;
    const int row0g = blockIdx.x * 64 + rt * 16;

    int mrow = row0g + (lane & 15); if (mrow >= rows) mrow = rows - 1;
    const _Float16* abase = A + (size_t)mrow * 128 + ((lane >> 4) << 3);

    v8f acc0 = {}, acc1 = {}, acc2 = {}, acc3 = {};
#pragma unroll
    for (int kt = 0; kt < 4; ++kt) {
        const _Float16* p = abase + kt * 32;
        v8h lo = *(const v8h*)(p);
        v8h hi = *(const v8h*)(p + 16);
        v16h a = __builtin_shufflevector(lo, hi, 0,1,2,3,4,5,6,7,8,9,10,11,12,13,14,15);
        acc0 = wmma_f16(a, load_b_frag(Wf, kt, ch * 4 + 0, lane), acc0);
        acc1 = wmma_f16(a, load_b_frag(Wf, kt, ch * 4 + 1, lane), acc1);
        acc2 = wmma_f16(a, load_b_frag(Wf, kt, ch * 4 + 2, lane), acc2);
        acc3 = wmma_f16(a, load_b_frag(Wf, kt, ch * 4 + 3, lane), acc3);
    }

    v8f accs[4] = {acc0, acc1, acc2, acc3};
    int mBase = row0g + ((lane >> 4) << 3);
    int nb = lane & 15;
#pragma unroll
    for (int j = 0; j < 4; ++j) {
        int col = (ch * 4 + j) * 16 + nb;
#pragma unroll
        for (int v = 0; v < 8; ++v) {
            int row = mBase + v;
            if (row < rows) {
                float f = accs[j][v];
                outSel[(size_t)row * 128 + col] = (_Float16)selu_f(f);
                if (AGG)
                    atomicMax(&aggEnc[(size_t)rcvr[row] * 128 + col], fenc(f));
            }
        }
    }
}

// ---------------- elementwise kernels ----------------
#define GIDX (blockIdx.x * blockDim.x + threadIdx.x)

// f32 W (K x 128) -> fragment-major f16:
// out[((kt*8+nt)*32+lane)*16+t] = W[(kt*32 + hi(lane)*8 + (t&7) + (t>>3)*16)*128 + nt*16 + lo(lane)]
__global__ void k_swizzle_w(const float* W, _Float16* out, int n /* = K*128 */) {
    int i = GIDX;
    if (i < n) {
        int t    = i & 15;
        int lane = (i >> 4) & 31;
        int blk  = i >> 9;                 // kt*8 + nt
        int kt = blk >> 3, nt = blk & 7;
        int k = kt * 32 + ((lane >> 4) << 3) + (t & 7) + ((t >> 3) << 4);
        int c = nt * 16 + (lane & 15);
        out[i] = (_Float16)W[(size_t)k * 128 + c];
    }
}
__global__ void k_split_x(const float* x, float* hs, float* ds, int n) {
    int i = GIDX; if (i < n) { hs[i] = x[2 * i]; ds[i] = x[2 * i + 1]; }
}
__global__ void k_copy_f32(const float* s, float* d, int n) {
    int i = GIDX; if (i < n) d[i] = s[i];
}
__global__ void k_fill_f32(float* d, float v, int n) {
    int i = GIDX; if (i < n) d[i] = v;
}
__global__ void k_fill_i32(int* d, int v, int n) {
    int i = GIDX; if (i < n) d[i] = v;
}
// g = selu([d_hat,d_star]) @ W(2x128); store gsel = f16(selu(g))
__global__ void k_node_in(const float* dh, const float* ds, const float* W,
                          _Float16* gsel, int n) {
    int i = GIDX;
    if (i < n * 128) {
        int node = i >> 7, c = i & 127;
        float s0 = selu_f(dh[node]), s1 = selu_f(ds[node]);
        gsel[i] = (_Float16)selu_f(s0 * W[c] + s1 * W[128 + c]);
    }
}
__global__ void k_edge_in(const float* qt, const float* qh, const float* W,
                          _Float16* zsel, int n) {
    int i = GIDX;
    if (i < n * 128) {
        int e = i >> 7, c = i & 127;
        float s0 = selu_f(qt[e]), s1 = selu_f(qh[e]);
        zsel[i] = (_Float16)selu_f(s0 * W[c] + s1 * W[128 + c]);
    }
}
// q = sign(dh)*(|dh|/r + zeta)^(1/1.852); optional segment-sum of q into d
__global__ void k_net_flow(const float* h, const float* r, const int* sndr,
                           const int* rcvr, float* q, float* dacc, int n) {
    int e = GIDX;
    if (e < n) {
        float dh  = h[sndr[e]] - h[rcvr[e]];
        float mag = __powf(fabsf(dh) / r[e] + ZETA_F, INV_HW);
        float qe  = dh > 0.f ? mag : (dh < 0.f ? -mag : 0.f);
        q[e] = qe;
        if (dacc) atomicAdd(&dacc[rcvr[e]], qe);
    }
}
__global__ void k_hl(const float* q, const float* r, float* hl, int n) {
    int e = GIDX;
    if (e < n) {
        float qe = q[e];
        float mag = __powf(fabsf(qe) + ZETA_F, HW_EXP);
        hl[e] = r[e] * (qe > 0.f ? mag : (qe < 0.f ? -mag : 0.f));
    }
}
// decode segment-max, isfinite->0, selu, f16; reset for next layer
__global__ void k_agg_decode(int* aggEnc, _Float16* aggsel, int n) {
    int i = GIDX;
    if (i < n) {
        int k = aggEnc[i];
        float a = fdec(k);
        if (k == IMIN || !(fabsf(a) < 3.0e38f)) a = 0.f;
        aggsel[i] = (_Float16)selu_f(a);
        aggEnc[i] = IMIN;
    }
}
// q_hat[:Eh] += f2 @ fl_w2 ; antisymmetrize into second half
__global__ void k_flow_update(const _Float16* f2, const float* fl2,
                              float* q_hat, int n) {
    int e = GIDX;
    if (e < n) {
        const _Float16* f = f2 + (size_t)e * 128;
        float acc = 0.f;
#pragma unroll 8
        for (int k = 0; k < 128; ++k) acc += (float)f[k] * fl2[k];
        float qn = q_hat[e] + acc;
        q_hat[e]      = qn;
        q_hat[e + EH] = -qn;
    }
}
__global__ void k_dsum(const float* q, const int* sndr, const int* rcvr,
                       float* d, int n) {
    int e = GIDX;
    if (e < n) {
        float v = q[e];
        atomicAdd(&d[rcvr[e]], v);
        atomicAdd(&d[sndr[e]], -v);
    }
}
__global__ void k_heads_edge(const float* h, const float* hl, const int* sndr,
                             const int* rcvr, int* updEnc, int n) {
    int e = GIDX;
    if (e < n) {
        float hs = h[sndr[e]];
        float cand = (hs != 0.f) ? hs - hl[e] : NEG_F;
        atomicMax(&updEnc[rcvr[e]], fenc(cand));
    }
}
__global__ void k_heads_node(float* h, const float* h_star, int* updEnc, int n) {
    int i = GIDX;
    if (i < n) {
        int k = updEnc[i];
        float upd = fdec(k);
        float hn;
        if (h_star[i] != 0.f) hn = h_star[i];
        else hn = (k != IMIN && upd > 0.5f * NEG_F) ? upd : h[i];
        h[i] = hn;
        updEnc[i] = IMIN;
    }
}

// ---------------- launcher ----------------
static inline size_t alignUp(size_t x) { return (x + 255) & ~(size_t)255; }

extern "C" void kernel_launch(void* const* d_in, const int* in_sizes, int n_in,
                              void* d_out, int out_size, void* d_ws, size_t ws_size,
                              hipStream_t stream) {
    (void)in_sizes; (void)n_in; (void)out_size; (void)ws_size;
    const float* x        = (const float*)d_in[0];
    const int*   eidx     = (const int*)  d_in[1];
    const float* r        = (const float*)d_in[2];
    const float* node_w   = (const float*)d_in[3];
    const float* edge_w   = (const float*)d_in[4];
    const float* fl_w0    = (const float*)d_in[5];
    const float* fl_w1    = (const float*)d_in[6];
    const float* fl_w2    = (const float*)d_in[7];
    const float* gcn_ew0  = (const float*)d_in[8];
    const float* gcn_ew1  = (const float*)d_in[9];
    const float* gcn_nw0  = (const float*)d_in[10];
    const float* gcn_nw1  = (const float*)d_in[11];
    const int* sndr = eidx;
    const int* rcvr = eidx + N_EDGES;
    float* h = (float*)d_out;                       // h_tilde lives in d_out

    // workspace layout
    char* p = (char*)d_ws;
    auto take = [&](size_t bytes) { char* q = p; p += alignUp(bytes); return q; };
    float*    h_star  = (float*)   take(N_NODES * 4);
    float*    d_star  = (float*)   take(N_NODES * 4);
    float*    q_hat   = (float*)   take(N_EDGES * 4);
    float*    q_tilde = (float*)   take(N_EDGES * 4);
    float*    d_hat   = (float*)   take(N_NODES * 4);
    float*    hl      = (float*)   take(N_EDGES * 4);
    int*      updEnc  = (int*)     take(N_NODES * 4);
    int*      aggEnc  = (int*)     take((size_t)N_NODES * 128 * 4);
    _Float16* gsel    = (_Float16*)take((size_t)N_NODES * 128 * 2);
    _Float16* aggsel  = (_Float16*)take((size_t)N_NODES * 128 * 2);
    _Float16* zsel    = (_Float16*)take((size_t)N_EDGES * 128 * 2);
    _Float16* tsel    = (_Float16*)take((size_t)N_EDGES * 128 * 2);
    _Float16* h_fl0   = (_Float16*)take(384 * 128 * 2);
    _Float16* h_fl1   = (_Float16*)take(128 * 128 * 2);
    _Float16* h_ew0   = (_Float16*)take((size_t)3 * 384 * 128 * 2);
    _Float16* h_ew1   = (_Float16*)take((size_t)3 * 128 * 128 * 2);
    _Float16* h_nw0   = (_Float16*)take((size_t)3 * 256 * 128 * 2);
    _Float16* h_nw1   = (_Float16*)take((size_t)3 * 128 * 128 * 2);
    _Float16* f2sel   = zsel;                       // reuse: free when fl1 runs

    const int T = 256;
    auto g1 = [&](int n) { return dim3((n + T - 1) / T); };
    dim3 gE64((N_EDGES + 63) / 64), gEh64((EH + 63) / 64), gN64((N_NODES + 63) / 64);

    // swizzle weights to fragment-major f16 (every call: deterministic)
    k_swizzle_w<<<g1(384*128), T, 0, stream>>>(fl_w0, h_fl0, 384*128);
    k_swizzle_w<<<g1(128*128), T, 0, stream>>>(fl_w1, h_fl1, 128*128);
    for (int l = 0; l < I_LAYERS; ++l) {
        k_swizzle_w<<<g1(384*128), T, 0, stream>>>(gcn_ew0 + (size_t)l*384*128,
                                                   h_ew0 + (size_t)l*384*128, 384*128);
        k_swizzle_w<<<g1(128*128), T, 0, stream>>>(gcn_ew1 + (size_t)l*128*128,
                                                   h_ew1 + (size_t)l*128*128, 128*128);
        k_swizzle_w<<<g1(256*128), T, 0, stream>>>(gcn_nw0 + (size_t)l*256*128,
                                                   h_nw0 + (size_t)l*256*128, 256*128);
        k_swizzle_w<<<g1(128*128), T, 0, stream>>>(gcn_nw1 + (size_t)l*128*128,
                                                   h_nw1 + (size_t)l*128*128, 128*128);
    }

    // init
    k_split_x<<<g1(N_NODES), T, 0, stream>>>(x, h_star, d_star, N_NODES);
    k_fill_i32<<<g1(N_NODES*128), T, 0, stream>>>(aggEnc, IMIN, N_NODES*128);
    k_fill_i32<<<g1(N_NODES), T, 0, stream>>>(updEnc, IMIN, N_NODES);
    k_fill_f32<<<g1(N_NODES), T, 0, stream>>>(d_hat, 0.f, N_NODES);
    k_net_flow<<<g1(N_EDGES), T, 0, stream>>>(h_star, r, sndr, rcvr, q_hat, d_hat, N_EDGES);
    k_copy_f32<<<g1(N_EDGES), T, 0, stream>>>(q_hat, q_tilde, N_EDGES);

    for (int it = 0; it < K_OUTER; ++it) {
        k_node_in<<<g1(N_NODES*128), T, 0, stream>>>(d_hat, d_star, node_w, gsel, N_NODES);
        k_edge_in<<<g1(N_EDGES*128), T, 0, stream>>>(q_tilde, q_hat, edge_w, zsel, N_EDGES);

        for (int l = 0; l < I_LAYERS; ++l) {
            // z' = selu([g_s|g_r|z] @ ew0)
            wmma_gemm_concat<3><<<gE64, T, 0, stream>>>(
                gsel, sndr, gsel, rcvr, zsel, nullptr,
                h_ew0 + (size_t)l*384*128, tsel, N_EDGES);
            // z = z' @ ew1 ; segment-max into aggEnc
            wmma_gemm_plain<true><<<gE64, T, 0, stream>>>(
                tsel, h_ew1 + (size_t)l*128*128, zsel, aggEnc, rcvr, N_EDGES);
            k_agg_decode<<<g1(N_NODES*128), T, 0, stream>>>(aggEnc, aggsel, N_NODES*128);
            // g' = selu([g|agg] @ nw0)
            wmma_gemm_concat<2><<<gN64, T, 0, stream>>>(
                gsel, nullptr, aggsel, nullptr, nullptr, nullptr,
                h_nw0 + (size_t)l*256*128, tsel, N_NODES);
            // g = g' @ nw1
            wmma_gemm_plain<false><<<gN64, T, 0, stream>>>(
                tsel, h_nw1 + (size_t)l*128*128, gsel, nullptr, nullptr, N_NODES);
        }

        // flow head (first half of edges suffices: q is antisymmetric)
        wmma_gemm_concat<3><<<gEh64, T, 0, stream>>>(
            gsel, sndr, gsel, rcvr, zsel, nullptr, h_fl0, tsel, EH);
        wmma_gemm_plain<false><<<gEh64, T, 0, stream>>>(
            tsel, h_fl1, f2sel, nullptr, nullptr, EH);
        k_flow_update<<<g1(EH), T, 0, stream>>>(f2sel, fl_w2, q_hat, EH);
        k_fill_f32<<<g1(N_NODES), T, 0, stream>>>(d_hat, 0.f, N_NODES);
        k_dsum<<<g1(EH), T, 0, stream>>>(q_hat, sndr, rcvr, d_hat, EH);

        // construct heads: J=9 scatter-max sweeps starting from h_star
        k_hl<<<g1(N_EDGES), T, 0, stream>>>(q_hat, r, hl, N_EDGES);
        k_copy_f32<<<g1(N_NODES), T, 0, stream>>>(h_star, h, N_NODES);
        for (int j = 0; j < J_HEADS; ++j) {
            k_heads_edge<<<g1(N_EDGES), T, 0, stream>>>(h, hl, sndr, rcvr, updEnc, N_EDGES);
            k_heads_node<<<g1(N_NODES), T, 0, stream>>>(h, h_star, updEnc, N_NODES);
        }
        // q_tilde from h_tilde (d_tilde is unused)
        k_net_flow<<<g1(N_EDGES), T, 0, stream>>>(h, r, sndr, rcvr, q_tilde, nullptr, N_EDGES);
    }
}
